// TensorProductConvLayer_240518168931
// MI455X (gfx1250) — matrix operand order repaired and checked by reading, not compiled, not run on hardware
//
#include <hip/hip_runtime.h>

#define N_NODES 10000
#define N_EDGES 100000

typedef __attribute__((ext_vector_type(16))) __bf16  v16bf;
typedef __attribute__((ext_vector_type(8)))  __bf16  v8bf;
typedef __attribute__((ext_vector_type(8)))  float   v8f;

__device__ __forceinline__ v16bf cat16(v8bf lo, v8bf hi) {
    return __builtin_shufflevector(lo, hi, 0,1,2,3,4,5,6,7,8,9,10,11,12,13,14,15);
}

__device__ __forceinline__ v16bf splat_bf(__bf16 s) {
    v16bf r;
    #pragma unroll
    for (int i = 0; i < 16; ++i) r[i] = s;
    return r;
}

__global__ void tpc_zero_kernel(float* __restrict__ out, int n) {
    int i = blockIdx.x * blockDim.x + threadIdx.x;
    if (i < n) out[i] = 0.0f;
}

// Workspace layout (__bf16):
//   W1t[4096]   : W1t[n*32+k]  <- W_fc1[k*128+n]
//   W2t[131072] : W2t[n*128+k] <- W_fc2[k*1024+n]
//   B2t[2560]   : bias B operand per acc-chain, B2t[chain*512 + n*32 + kk]
//                 chain0: kk<16 -> b2[blk0][kk,n], kk>=16 -> b2[blk3][kk-16,n]
//                 chain1: kk<16 -> b2[blk1][kk,n], else 0
//                 chain2..4: kk<16 -> b2[blk2][kk,n], else 0
__global__ void tpc_convert_kernel(const float* __restrict__ W1,
                                   const float* __restrict__ W2,
                                   const float* __restrict__ b2,
                                   __bf16* __restrict__ W1t,
                                   __bf16* __restrict__ W2t,
                                   __bf16* __restrict__ B2t) {
    int i = blockIdx.x * blockDim.x + threadIdx.x;
    if (i < 4096) {
        int n = i >> 5, k = i & 31;
        W1t[i] = (__bf16)W1[k * 128 + n];
    } else if (i < 4096 + 131072) {
        int j = i - 4096;
        int n = j >> 7, k = j & 127;
        W2t[j] = (__bf16)W2[k * 1024 + n];
    } else if (i < 4096 + 131072 + 2560) {
        int j = i - (4096 + 131072);
        int chain = j >> 9, r = j & 511, n = r >> 5, kk = r & 31;
        float v;
        if (chain == 0) v = (kk < 16) ? b2[kk * 16 + n] : b2[768 + (kk - 16) * 16 + n];
        else if (chain == 1) v = (kk < 16) ? b2[256 + kk * 16 + n] : 0.0f;
        else v = (kk < 16) ? b2[512 + kk * 16 + n] : 0.0f;
        B2t[j] = (__bf16)v;
    }
}

__global__ __launch_bounds__(256)
void tpconv_kernel(const float* __restrict__ node_attr,
                   const int*   __restrict__ edge_index,
                   const float* __restrict__ edge_attr,
                   const float* __restrict__ edge_feat,
                   const float* __restrict__ b_fc1,
                   const __bf16* __restrict__ W1t,
                   const __bf16* __restrict__ W2t,
                   const __bf16* __restrict__ B2t,
                   float* __restrict__ out) {
    __shared__ __align__(16) __bf16 hB[32][128];    // relu(fc1), bf16
    __shared__ __align__(16) __bf16 sc[6][32][16];  // per-edge scale rows, bf16
    __shared__ float acc[5][32][16];                // chain accumulators
    __shared__ float shv[32][3];
    __shared__ int   srcs[32];

    const int t     = threadIdx.x;
    const int wave  = t >> 5;                      // wave32
    const int lane  = t & 31;
    const int nn    = lane & 15;                   // B/C/D column within tile
    const int m     = lane & 15;                   // A row within tile
    const int kbase = (lane < 16) ? 0 : 8;         // A K-group (16-bit layout)
    const int kb16  = (lane < 16) ? 0 : 8;         // B K-group
    const int rwof  = (lane < 16) ? 0 : 8;         // C/D row offset
    const int ebase = blockIdx.x * 32;

    const float INV3c = 0.57735026918962576f;      // 1/sqrt(3)
    const float SC2   = 0.017677669529663687f;     // (1/sqrt(32)) * (1/10)

    for (int i = t; i < 5 * 32 * 16; i += 256) ((float*)acc)[i] = 0.0f;

    // ---- prologue: per-edge scale vectors (A-operand-friendly bf16 rows) ----
    {
        int e  = t >> 3;                 // 0..31
        int u0 = (t & 7) * 2;            // 2 u values per thread
        int eg = ebase + e;
        int dst = edge_index[N_EDGES + eg];
        float sh0 = edge_attr[eg * 4 + 0];
        float sh1 = edge_attr[eg * 4 + 1];
        float sh2 = edge_attr[eg * 4 + 2];
        float sh3 = edge_attr[eg * 4 + 3];
        if ((t & 7) == 0) {
            srcs[e] = edge_index[eg];
            shv[e][0] = sh1; shv[e][1] = sh2; shv[e][2] = sh3;
        }
        const float* x = node_attr + (size_t)dst * 64;
        #pragma unroll
        for (int du = 0; du < 2; ++du) {
            int u = u0 + du;
            float xs  = x[u];
            float xv0 = x[16 + 3 * u + 0];
            float xv1 = x[16 + 3 * u + 1];
            float xv2 = x[16 + 3 * u + 2];
            sc[0][e][u] = (__bf16)(xs * sh0);                                    // a    (w1)
            sc[1][e][u] = (__bf16)(INV3c * (xv0 * sh1 + xv1 * sh2 + xv2 * sh3)); // d/√3 (w4)
            sc[2][e][u] = (__bf16)xs;                                            //      (w2)
            sc[3][e][u] = (__bf16)(xv0 * sh0);                                   //      (w3,k0)
            sc[4][e][u] = (__bf16)(xv1 * sh0);                                   //      (w3,k1)
            sc[5][e][u] = (__bf16)(xv2 * sh0);                                   //      (w3,k2)
        }
    }

    // ---- fc1: h = relu(ef @ W1 + b1) -> bf16 LDS; col tile `wave`, 2 row tiles ----
    {
        const __bf16* bp = W1t + ((wave * 16 + nn) * 32 + kb16);
        v16bf bv = cat16(*(const v8bf*)bp, *(const v8bf*)(bp + 16));
        float bias = b_fc1[wave * 16 + nn];
        #pragma unroll
        for (int rt = 0; rt < 2; ++rt) {
            const float* ef = edge_feat + (size_t)(ebase + rt * 16 + m) * 32;
            const float4* p0 = (const float4*)(ef + kbase);
            const float4* p1 = (const float4*)(ef + kbase + 16);
            float4 a0 = p0[0], a1 = p0[1], a2 = p1[0], a3 = p1[1];
            v16bf av;
            av[0]=(__bf16)a0.x; av[1]=(__bf16)a0.y; av[2]=(__bf16)a0.z; av[3]=(__bf16)a0.w;
            av[4]=(__bf16)a1.x; av[5]=(__bf16)a1.y; av[6]=(__bf16)a1.z; av[7]=(__bf16)a1.w;
            av[8]=(__bf16)a2.x; av[9]=(__bf16)a2.y; av[10]=(__bf16)a2.z; av[11]=(__bf16)a2.w;
            av[12]=(__bf16)a3.x; av[13]=(__bf16)a3.y; av[14]=(__bf16)a3.z; av[15]=(__bf16)a3.w;
            v8f c;
            #pragma unroll
            for (int r = 0; r < 8; ++r) c[r] = bias;
            c = __builtin_amdgcn_wmma_f32_16x16x32_bf16(false, av, false, bv,
                                                        (short)0, c, false, false);
            #pragma unroll
            for (int r = 0; r < 8; ++r)
                hB[rt * 16 + r + rwof][wave * 16 + nn] = (__bf16)fmaxf(c[r], 0.0f);
        }
    }

    __syncthreads();

    // ---- fused fc2 + tensor product: row-scaled WMMA accumulation ----
    // 24 uniform subchains = (virtual chain 0..5) x (kc 0..3), 16 u-steps each;
    // exactly 3 subchains per wave. vc0 (w1) and vc1 (w4) both reduce into acc0.
    // Bias folded as one extra WMMA on the kc==0 subchain of vc0,2,3,4,5.
    for (int js = wave; js < 24; js += 8) {
        int vc = js >> 2;
        int kc = js & 3;
        int b     = (vc == 0) ? 0 : (vc == 1) ? 3 : (vc == 2) ? 1 : 2;
        int slot  = vc;
        int accid = (vc == 0) ? 0 : vc - 1;

        const v8bf* q0 = (const v8bf*)&hB[m][kc * 32 + kbase];
        const v8bf* q1 = (const v8bf*)&hB[m][kc * 32 + kbase + 16];
        v16bf hv0 = cat16(q0[0], q1[0]);
        const v8bf* q2 = (const v8bf*)&hB[16 + m][kc * 32 + kbase];
        const v8bf* q3 = (const v8bf*)&hB[16 + m][kc * 32 + kbase + 16];
        v16bf hv1 = cat16(q2[0], q3[0]);

        v16bf sv0 = *(const v16bf*)&sc[slot][m][0];
        v16bf sv1 = *(const v16bf*)&sc[slot][16 + m][0];

        v8f c0, c1;
        #pragma unroll
        for (int r = 0; r < 8; ++r) { c0[r] = 0.0f; c1[r] = 0.0f; }

        const __bf16* bbase = W2t + ((size_t)(b * 256 + nn)) * 128 + (kc * 32 + kb16);
        #pragma unroll
        for (int u = 0; u < 16; ++u) {
            v16bf a0 = hv0 * splat_bf(sv0[u]);
            v16bf a1 = hv1 * splat_bf(sv1[u]);
            const __bf16* bp = bbase + (size_t)u * 2048;   // 16 cols * 128 stride
            v16bf bv = cat16(*(const v8bf*)bp, *(const v8bf*)(bp + 16));
            c0 = __builtin_amdgcn_wmma_f32_16x16x32_bf16(false, a0, false, bv,
                                                         (short)0, c0, false, false);
            c1 = __builtin_amdgcn_wmma_f32_16x16x32_bf16(false, a1, false, bv,
                                                         (short)0, c1, false, false);
        }

        if (kc == 0 && vc != 1) {  // bias WMMA: A = scale rows, B = packed b_fc2
            int s2 = (vc == 0) ? 1 : slot;    // upper-K B rows are zero unless vc0
            v16bf a0 = cat16(*(const v8bf*)&sc[slot][m][kbase],
                             *(const v8bf*)&sc[s2][m][kbase]);
            v16bf a1 = cat16(*(const v8bf*)&sc[slot][16 + m][kbase],
                             *(const v8bf*)&sc[s2][16 + m][kbase]);
            const __bf16* bp = B2t + accid * 512 + nn * 32 + kb16;
            v16bf bv = cat16(*(const v8bf*)bp, *(const v8bf*)(bp + 16));
            c0 = __builtin_amdgcn_wmma_f32_16x16x32_bf16(false, a0, false, bv,
                                                         (short)0, c0, false, false);
            c1 = __builtin_amdgcn_wmma_f32_16x16x32_bf16(false, a1, false, bv,
                                                         (short)0, c1, false, false);
        }

        #pragma unroll
        for (int r = 0; r < 8; ++r) {
            atomicAdd(&acc[accid][r + rwof][nn], c0[r]);
            atomicAdd(&acc[accid][16 + r + rwof][nn], c1[r]);
        }
    }

    __syncthreads();

    // ---- epilogue: combine chains, apply sh_v outer product, scatter-add ----
    {
        int e0 = t >> 4, w = t & 15;
        #pragma unroll
        for (int pass = 0; pass < 2; ++pass) {
            int e = e0 + pass * 16;
            float outs = acc[0][e][w] * SC2;
            float s2v  = acc[1][e][w];
            float o0 = (s2v * shv[e][0] + acc[2][e][w]) * SC2;
            float o1 = (s2v * shv[e][1] + acc[3][e][w]) * SC2;
            float o2 = (s2v * shv[e][2] + acc[4][e][w]) * SC2;
            float* op = out + (size_t)srcs[e] * 64;
            atomicAdd(op + w, outs);
            atomicAdd(op + 16 + w * 3 + 0, o0);
            atomicAdd(op + 16 + w * 3 + 1, o1);
            atomicAdd(op + 16 + w * 3 + 2, o2);
        }
    }
}

extern "C" void kernel_launch(void* const* d_in, const int* in_sizes, int n_in,
                              void* d_out, int out_size, void* d_ws, size_t ws_size,
                              hipStream_t stream) {
    const float* node_attr  = (const float*)d_in[0];
    const int*   edge_index = (const int*)  d_in[1];
    const float* edge_attr  = (const float*)d_in[2];
    const float* edge_feat  = (const float*)d_in[3];
    const float* W_fc1      = (const float*)d_in[4];
    const float* b_fc1      = (const float*)d_in[5];
    const float* W_fc2      = (const float*)d_in[6];
    const float* b_fc2      = (const float*)d_in[7];
    float* out = (float*)d_out;

    __bf16* W1t = (__bf16*)d_ws;
    __bf16* W2t = W1t + 4096;
    __bf16* B2t = W2t + 131072;

    tpc_zero_kernel<<<(N_NODES * 64 + 255) / 256, 256, 0, stream>>>(out, N_NODES * 64);
    tpc_convert_kernel<<<(4096 + 131072 + 2560 + 255) / 256, 256, 0, stream>>>(
        W_fc1, W_fc2, b_fc2, W1t, W2t, B2t);
    tpconv_kernel<<<N_EDGES / 32, 256, 0, stream>>>(node_attr, edge_index, edge_attr,
                                                    edge_feat, b_fc1,
                                                    W1t, W2t, B2t, out);
    (void)in_sizes; (void)n_in; (void)out_size; (void)ws_size;
}
